// GraphAttentionLayer_1726576855319
// MI455X (gfx1250) — matrix operand order repaired and checked by reading, not compiled, not run on hardware
//
#include <hip/hip_runtime.h>
#include <hip/hip_bf16.h>

typedef __bf16   v16bf __attribute__((ext_vector_type(16)));
typedef float    v8f   __attribute__((ext_vector_type(8)));
typedef __bf16   bf2   __attribute__((ext_vector_type(2)));
typedef unsigned u32x4 __attribute__((ext_vector_type(4)));
typedef unsigned u32x8 __attribute__((ext_vector_type(8)));

#define N_NODES 8192
#define F_IN    512
#define F_OUT   64
#define ALPHA   0.2f
#define JB      256

__device__ __forceinline__ float lrelu(float x) { return x > 0.f ? x : ALPHA * x; }

// ---------------------------------------------------------------------------
// TDM 2D tile load: copies `tile_rows` rows x 512 B (256 bf16) from a row-major
// bf16 tensor with row stride N_NODES elements into LDS, inserting 16 B of pad
// after every 512 B stored (matches the [JB+8] LDS row layout).
// D# group 0: count=1 | lds_addr | global_addr | type=2.
// D# group 1: data_size=2B, pad_enable, pad_interval=128 DW (512 B),
//             pad_amount=4 DW (16 B), tensor_dim0=8192, tile_dim0=256,
//             tensor_dim0_stride=8192.
// Issued once per wave (EXEC-independent); tracked by TENSORcnt.
// ---------------------------------------------------------------------------
__device__ __forceinline__ void tdm_load_tile(unsigned lds_byte_addr,
                                              const __bf16* gsrc,
                                              unsigned tile_rows)
{
    unsigned long ga = (unsigned long)gsrc;
    u32x4 g0;
    g0.x = 1u;                                         // count=1 (valid user D#)
    g0.y = lds_byte_addr;                              // LDS dest (bytes)
    g0.z = (unsigned)ga;                               // global_addr[31:0]
    g0.w = (unsigned)((ga >> 32) & 0x01FFFFFFu)        // global_addr[56:32]
         | 0x80000000u;                                // type=2 ("image")
    u32x8 g1;
    g1[0] = (1u << 16)                                 // data_size = 2 bytes
          | (1u << 20)                                 // pad_enable
          | (6u << 22)                                 // pad_interval: 128 DWORDs
          | (3u << 25);                                // pad_amount: 4 DWORDs
    g1[1] = (N_NODES & 0xFFFFu) << 16;                 // tensor_dim0[15:0]
    g1[2] = ((N_NODES >> 16) & 0xFFFFu)                // tensor_dim0[31:16]
          | (tile_rows << 16);                         // tensor_dim1[15:0]
    g1[3] = ((unsigned)JB << 16);                      // tile_dim0 = 256
    g1[4] = tile_rows;                                 // tile_dim1 (tile_dim2=0)
    g1[5] = (unsigned)N_NODES;                         // tensor_dim0_stride lo32
    g1[6] = 0u;
    g1[7] = 0u;
    asm volatile("tensor_load_to_lds %0, %1" :: "s"(g0), "s"(g1) : "memory");
}

// ---------------------------------------------------------------------------
// Kernel 1: hT = bf16(input @ W) stored TRANSPOSED as hT[64][8192].
// One wave computes one 16x16 tile via v_wmma_f32_16x16x32_bf16.
// C layout (lane L, VGPR r): M = (L/16)*8 + r, N = L%16  ->  8 consecutive
// m-addresses per lane in hT  ->  single global_store_b128.
// ---------------------------------------------------------------------------
__global__ __launch_bounds__(256) void k_gemm_h(const float* __restrict__ input,
                                                const float* __restrict__ W,
                                                __bf16* __restrict__ hT)
{
    const int lane = threadIdx.x & 31;
    const int w    = threadIdx.x >> 5;
    const int gw   = blockIdx.x * 8 + w;     // 0..2047
    const int tm   = gw >> 2;                // 0..511 (row tile)
    const int tn   = gw & 3;                 // 0..3   (col tile)
    const int half = lane >> 4;
    const int l16  = lane & 15;

    v8f c = {};
    for (int k0 = 0; k0 < F_IN; k0 += 32) {
        v16bf a, b;
#pragma unroll
        for (int v = 0; v < 8; ++v) {
            const int ka = k0 + (v >> 2) * 16 + half * 8 + (v & 3) * 2;
            const float2 pa = *(const float2*)(input + (size_t)(tm * 16 + l16) * F_IN + ka);
            a[2 * v]     = (__bf16)pa.x;
            a[2 * v + 1] = (__bf16)pa.y;
            const int kb = k0 + (v >> 2) * 16 + half * 8 + (v & 3) * 2;
            const int n  = tn * 16 + l16;
            b[2 * v]     = (__bf16)W[(size_t)kb * F_OUT + n];
            b[2 * v + 1] = (__bf16)W[(size_t)(kb + 1) * F_OUT + n];
        }
        c = __builtin_amdgcn_wmma_f32_16x16x32_bf16(false, a, false, b, (short)0, c, false, false);
    }
    uint4 st;
    {
        bf2 t0; t0.x = (__bf16)c[0]; t0.y = (__bf16)c[1]; st.x = __builtin_bit_cast(unsigned, t0);
        bf2 t1; t1.x = (__bf16)c[2]; t1.y = (__bf16)c[3]; st.y = __builtin_bit_cast(unsigned, t1);
        bf2 t2; t2.x = (__bf16)c[4]; t2.y = (__bf16)c[5]; st.z = __builtin_bit_cast(unsigned, t2);
        bf2 t3; t3.x = (__bf16)c[6]; t3.y = (__bf16)c[7]; st.w = __builtin_bit_cast(unsigned, t3);
    }
    const int n  = tn * 16 + l16;
    const int m0 = tm * 16 + half * 8;
    *(uint4*)(hT + (size_t)n * N_NODES + m0) = st;
}

// ---------------------------------------------------------------------------
// Kernel 2: f1 = h@a1, f2 = h@a2 from transposed hT (coalesced across rows).
// ---------------------------------------------------------------------------
__global__ __launch_bounds__(256) void k_f12(const __bf16* __restrict__ hT,
                                             const float* __restrict__ a,
                                             float* __restrict__ f1,
                                             float* __restrict__ f2)
{
    const int i = blockIdx.x * 256 + threadIdx.x;
    float s1 = 0.f, s2 = 0.f;
#pragma unroll 8
    for (int c = 0; c < F_OUT; ++c) {
        const float hv = (float)hT[(size_t)c * N_NODES + i];
        s1 += hv * a[c];
        s2 += hv * a[F_OUT + c];
    }
    f1[i] = s1;
    f2[i] = s2;
}

// ---------------------------------------------------------------------------
// Kernel 3: gmax = max_j f2[j]  (upper bound for single-pass stable softmax)
// ---------------------------------------------------------------------------
__global__ __launch_bounds__(256) void k_gmax(const float* __restrict__ f2,
                                              float* __restrict__ gmax)
{
    __shared__ float red[256];
    float m = -1e30f;
    for (int i = threadIdx.x; i < N_NODES; i += 256) m = fmaxf(m, f2[i]);
    red[threadIdx.x] = m;
    __syncthreads();
    for (int s = 128; s > 0; s >>= 1) {
        if (threadIdx.x < s) red[threadIdx.x] = fmaxf(red[threadIdx.x], red[threadIdx.x + s]);
        __syncthreads();
    }
    if (threadIdx.x == 0) gmax[0] = red[0];
}

// ---------------------------------------------------------------------------
// Kernel 4: fused masked-softmax attention + P@h + ELU.
// One block (8 waves) per 16-row tile. Per 256-column J block:
//   - each wave issues a TDM tensor_load_to_lds for its 8 rows of the h^T tile
//     (async, overlaps the adj stream)
//   - threads stream adj with NT loads, compute p = mask ? exp(e - M_i) : 0,
//     accumulate fp32 row sums, store bf16 p tile to LDS
//   - s_wait_tensorcnt 0 + barrier, then each wave WMMAs its 32-wide K chunk
//     into 4 fp32 16x16 accumulators (16x64 partial).
// Final: cross-wave LDS reduction, softmax normalize, ELU.
// ---------------------------------------------------------------------------
__global__ __launch_bounds__(256) void k_attn(const int* __restrict__ adj,
                                              const __bf16* __restrict__ hT,
                                              const float* __restrict__ f1,
                                              const float* __restrict__ f2,
                                              const float* __restrict__ gmaxp,
                                              float* __restrict__ out)
{
    __shared__ __bf16 p_lds[16][JB + 8];        // attention weights tile (bf16)
    __shared__ __bf16 h_lds[F_OUT][JB + 8];     // h^T tile; pad matches TDM pad
    __shared__ float  partial[8][16][F_OUT];    // per-wave fp32 partials
    __shared__ float  rowsum[16];
    __shared__ float  f1s[16], Ms[16];

    const int tid  = threadIdx.x;
    const int lane = tid & 31;
    const int w    = tid >> 5;
    const int half = lane >> 4;
    const int l16  = lane & 15;
    const int i0   = blockIdx.x * 16;

    if (tid < 16) {
        rowsum[tid] = 0.f;
        const float fv = f1[i0 + tid];
        f1s[tid] = fv;
        Ms[tid]  = lrelu(fv + gmaxp[0]);        // per-row logit upper bound
    }
    __syncthreads();

    float rs[16];
#pragma unroll
    for (int r = 0; r < 16; ++r) rs[r] = 0.f;

    v8f acc[4];
#pragma unroll
    for (int nt = 0; nt < 4; ++nt) { v8f z = {}; acc[nt] = z; }

    const unsigned lds_dst = (unsigned)(size_t)&h_lds[w * 8][0];

    for (int jb = 0; jb < N_NODES; jb += JB) {
        // ---- async: this wave's 8 rows of h^T tile via Tensor Data Mover ---
        tdm_load_tile(lds_dst, hT + (size_t)(w * 8) * N_NODES + jb, 8u);

        // ---- masked exp weights: coalesced NT adj stream -------------------
        const float f2j = f2[jb + tid];
#pragma unroll
        for (int r = 0; r < 16; ++r) {
            const int av = __builtin_nontemporal_load(
                adj + (size_t)(i0 + r) * N_NODES + jb + tid);
            const float e = lrelu(f1s[r] + f2j);
            const float p = (av > 0) ? __expf(e - Ms[r]) : 0.f;
            rs[r] += p;
            p_lds[r][tid] = (__bf16)p;
        }

        __builtin_amdgcn_s_wait_tensorcnt(0);
        __syncthreads();

        // ---- each wave: WMMA its 32-wide K chunk into 16x64 partial --------
        const int kb = w * 32;
        v16bf a;
#pragma unroll
        for (int v = 0; v < 8; ++v) {
            const int kk = kb + (v >> 2) * 16 + half * 8 + (v & 3) * 2;
            const unsigned u = *(const unsigned*)&p_lds[l16][kk];
            const bf2 pr = __builtin_bit_cast(bf2, u);
            a[2 * v] = pr.x; a[2 * v + 1] = pr.y;
        }
#pragma unroll
        for (int nt = 0; nt < 4; ++nt) {
            v16bf b;
            const int n = nt * 16 + l16;
#pragma unroll
            for (int v = 0; v < 8; ++v) {
                const int kk = kb + (v >> 2) * 16 + half * 8 + (v & 3) * 2;
                const unsigned u = *(const unsigned*)&h_lds[n][kk];
                const bf2 pr = __builtin_bit_cast(bf2, u);
                b[2 * v] = pr.x; b[2 * v + 1] = pr.y;
            }
            acc[nt] = __builtin_amdgcn_wmma_f32_16x16x32_bf16(
                false, a, false, b, (short)0, acc[nt], false, false);
        }
        __syncthreads();   // all reads of p_lds/h_lds done before next TDM/stores
    }

    // ---- cross-wave reduction, softmax normalize, ELU ----------------------
#pragma unroll
    for (int nt = 0; nt < 4; ++nt)
#pragma unroll
        for (int r = 0; r < 8; ++r)
            partial[w][half * 8 + r][nt * 16 + l16] = acc[nt][r];
#pragma unroll
    for (int r = 0; r < 16; ++r) atomicAdd(&rowsum[r], rs[r]);
    __syncthreads();

#pragma unroll
    for (int q = 0; q < 4; ++q) {
        const int idx = q * 256 + tid;  // 0..1023 over 16x64 outputs
        const int m = idx >> 6, c = idx & 63;
        float s = 0.f;
#pragma unroll
        for (int ww = 0; ww < 8; ++ww) s += partial[ww][m][c];
        const float denom = rowsum[m];
        const float val = (denom > 0.f) ? s / denom : 0.f;
        out[(size_t)(i0 + m) * F_OUT + c] = val > 0.f ? val : __expf(val) - 1.f;
    }
}

// ---------------------------------------------------------------------------
extern "C" void kernel_launch(void* const* d_in, const int* in_sizes, int n_in,
                              void* d_out, int out_size, void* d_ws, size_t ws_size,
                              hipStream_t stream)
{
    const float* input = (const float*)d_in[0];   // 8192 x 512
    const int*   adj   = (const int*)d_in[1];     // 8192 x 8192
    const float* W     = (const float*)d_in[2];   // 512 x 64
    const float* a     = (const float*)d_in[3];   // 128 x 1
    float*       out   = (float*)d_out;           // 8192 x 64

    char* ws = (char*)d_ws;
    __bf16* hT   = (__bf16*)ws;                                   // 1 MB (64 x 8192, transposed)
    float*  f1   = (float*)(ws + (size_t)N_NODES * F_OUT * 2);    // 32 KB
    float*  f2   = f1 + N_NODES;                                  // 32 KB
    float*  gmax = f2 + N_NODES;                                  // 4 B

    k_gemm_h<<<256, 256, 0, stream>>>(input, W, hT);
    k_f12<<<N_NODES / 256, 256, 0, stream>>>(hT, a, f1, f2);
    k_gmax<<<1, 256, 0, stream>>>(f2, gmax);
    k_attn<<<N_NODES / 16, 256, 0, stream>>>(adj, hT, f1, f2, gmax, out);
}
